// GNN_58789512348198
// MI455X (gfx1250) — matrix operand order repaired and checked by reading, not compiled, not run on hardware
//
#include <hip/hip_runtime.h>

// ---------------------------------------------------------------------------
// GraphSAGE 2-layer forward on MI455X (gfx1250, wave32, WMMA bf16).
// HBM-bound (~398 MB single-use f32 inputs, ~10 GFLOP matmul):
//   * stream raw features once with NON-TEMPORAL loads (keep 192MB L2 for the
//     reused bf16 activation/weight matrices),
//   * all GEMM operands bf16, f32 accumulate via v_wmma_f32_16x16x32_bf16,
//   * per-block weight slice staged in LDS (bank-conflict-free padded pitch),
//     read back with ds_load_b128 -> 8x less L2 traffic for B,
//   * K-loop manually unrolled 2x with named ping-pong A fragments (no
//     register copies -> no WMMA->VALU WAR hazard NOPs) and both halves' B
//     fragments issued up front so WMMAs overlap outstanding ds/global loads,
//   * relu fused into the accumulator store; final f32 outputs stored NT.
// ---------------------------------------------------------------------------

typedef __attribute__((ext_vector_type(16))) __bf16 v16bf;
typedef __attribute__((ext_vector_type(8)))  float  v8f;
typedef __attribute__((ext_vector_type(4)))  unsigned int v4u;

union FragAB {
    v16bf bf;
    v4u   u[2];
};

#define FANOUT 10

// ---------------------------------------------------------------------------
// Pack weights: W [K0, N] f32 row-major -> Wt [N, Kp] bf16 (K zero-padded).
// ---------------------------------------------------------------------------
__global__ void pack_w_kernel(const float* __restrict__ W, __bf16* __restrict__ Wt,
                              int K0, int N, int Kp) {
    const int n = blockIdx.x;
    __bf16* orow = Wt + (size_t)n * Kp;
    for (int k = threadIdx.x; k < Kp; k += blockDim.x) {
        float v = (k < K0) ? W[(size_t)k * N + n] : 0.0f;
        orow[k] = (__bf16)v;
    }
}

// ---------------------------------------------------------------------------
// Layer-0 aggregation: out[row, 0:d) = self; out[row, d:2d) = mean of FANOUT
// neighbor rows; out[row, 2d:Kp) = 0.  f32 in -> bf16 out.  One block per row.
// Raw features are single-use: non-temporal loads keep them out of L2.
// ---------------------------------------------------------------------------
__global__ void agg_concat_f32_kernel(const float* __restrict__ self,
                                      const float* __restrict__ neigh,
                                      __bf16* __restrict__ out,
                                      int d, int Kp) {
    const int row = blockIdx.x;
    const float inv = 1.0f / (float)FANOUT;
    const float* srow  = self  + (size_t)row * d;
    const float* nbase = neigh + (size_t)row * FANOUT * d;
    __bf16* orow = out + (size_t)row * Kp;
    for (int c = threadIdx.x; c < Kp; c += blockDim.x) {
        float v = 0.0f;
        if (c < d) {
            v = __builtin_nontemporal_load(srow + c);
        } else if (c < 2 * d) {
            const int cc = c - d;
            float s = 0.0f;
#pragma unroll
            for (int j = 0; j < FANOUT; ++j)
                s += __builtin_nontemporal_load(nbase + (size_t)j * d + cc);
            v = s * inv;
        }
        orow[c] = (__bf16)v;
    }
}

// Same, but bf16 in (layer-1 aggregates the L2-hot bf16 hidden activations,
// so these use regular temporal loads).
__global__ void agg_concat_bf16_kernel(const __bf16* __restrict__ self,
                                       const __bf16* __restrict__ neigh,
                                       __bf16* __restrict__ out,
                                       int d, int Kp) {
    const int row = blockIdx.x;
    const float inv = 1.0f / (float)FANOUT;
    const __bf16* srow  = self  + (size_t)row * d;
    const __bf16* nbase = neigh + (size_t)row * FANOUT * d;
    __bf16* orow = out + (size_t)row * Kp;
    for (int c = threadIdx.x; c < Kp; c += blockDim.x) {
        float v = 0.0f;
        if (c < d) {
            v = (float)srow[c];
        } else if (c < 2 * d) {
            const int cc = c - d;
            float s = 0.0f;
#pragma unroll
            for (int j = 0; j < FANOUT; ++j) s += (float)nbase[(size_t)j * d + cc];
            v = s * inv;
        }
        orow[c] = (__bf16)v;
    }
}

// ---------------------------------------------------------------------------
// GEMM + relu:  C[M,N] = relu(A[M,K] @ Bt[N,K]^T), bf16 operands, f32 acc.
// Block = 256 threads = 8 waves; wave w owns rows [blk.x*128 + w*16, +16),
// cols [blk.y*64, +64).  Weight slice staged once in LDS (pitch K+8 bf16 =>
// conflict-free fragment reads).  K-loop unrolled 2x with ping-pong A frags.
// Requires M % 128 == 0, N % 64 == 0, K % 64 == 0 (true at all call sites).
// ---------------------------------------------------------------------------
template <typename T, bool NT_STORE>
__global__ __launch_bounds__(256) void
gemm_bf16_relu_kernel(const __bf16* __restrict__ A,
                      const __bf16* __restrict__ Bt,
                      T* __restrict__ C,
                      int M, int N, int K) {
    extern __shared__ __bf16 ldsB[];          // [64][K + 8]
    const int pitch = K + 8;                  // +16B: distinct LDS banks/lane
    const int tid   = threadIdx.x;
    const int lane  = tid & 31;
    const int wave  = tid >> 5;
    const int m0    = blockIdx.x * 128 + wave * 16;
    const int n0    = blockIdx.y * 64;
    const int nh    = lane & 15;              // row (A) / col (B,C) within tile
    const int kgrp  = lane >> 4;              // K-half selector

    // ---- Cooperative stage of the 64 x K weight slice into LDS ----
    {
        const int chunksPerRow = K >> 3;      // b128 chunks (8 bf16)
        const int totalChunks  = 64 * chunksPerRow;
        for (int idx = tid; idx < totalChunks; idx += 256) {
            const int r  = idx / chunksPerRow;
            const int kc = (idx - r * chunksPerRow) << 3;
            *(v4u*)(&ldsB[r * pitch + kc]) =
                *(const v4u*)(Bt + (size_t)(n0 + r) * K + kc);
        }
    }
    __syncthreads();

    // A fragment: lane holds row m0+nh, K chunks at kgrp*8 and kgrp*8+16.
    const __bf16* aPtr = A + (size_t)(m0 + nh) * K + kgrp * 8;
    // B fragments from LDS: lane holds col nt*16+nh, K chunks at kgrp*8 and kgrp*8+16.
    const __bf16* lb0 = ldsB + ( 0 + nh) * pitch + kgrp * 8;
    const __bf16* lb1 = ldsB + (16 + nh) * pitch + kgrp * 8;
    const __bf16* lb2 = ldsB + (32 + nh) * pitch + kgrp * 8;
    const __bf16* lb3 = ldsB + (48 + nh) * pitch + kgrp * 8;

    v8f acc0 = {}, acc1 = {}, acc2 = {}, acc3 = {};

    FragAB fa0, fa1;                           // ping-pong A fragments
    fa0.u[0] = *(const v4u*)(aPtr);
    fa0.u[1] = *(const v4u*)(aPtr + 16);

#pragma unroll 1
    for (int k0 = 0; k0 < K; k0 += 64) {
        const int k1 = k0 + 32;

        // A fragment for the 2nd half of this iteration (always in range:
        // K % 64 == 0 so k1 <= K - 32).
        fa1.u[0] = *(const v4u*)(aPtr + k1);
        fa1.u[1] = *(const v4u*)(aPtr + k1 + 16);

        // Issue BOTH halves' B fragments now: the first WMMA group only has
        // to drain half the outstanding ds loads.
        FragAB b00, b01, b02, b03, b10, b11, b12, b13;
        b00.u[0] = *(const v4u*)(lb0 + k0);  b00.u[1] = *(const v4u*)(lb0 + k0 + 16);
        b01.u[0] = *(const v4u*)(lb1 + k0);  b01.u[1] = *(const v4u*)(lb1 + k0 + 16);
        b02.u[0] = *(const v4u*)(lb2 + k0);  b02.u[1] = *(const v4u*)(lb2 + k0 + 16);
        b03.u[0] = *(const v4u*)(lb3 + k0);  b03.u[1] = *(const v4u*)(lb3 + k0 + 16);
        b10.u[0] = *(const v4u*)(lb0 + k1);  b10.u[1] = *(const v4u*)(lb0 + k1 + 16);
        b11.u[0] = *(const v4u*)(lb1 + k1);  b11.u[1] = *(const v4u*)(lb1 + k1 + 16);
        b12.u[0] = *(const v4u*)(lb2 + k1);  b12.u[1] = *(const v4u*)(lb2 + k1 + 16);
        b13.u[0] = *(const v4u*)(lb3 + k1);  b13.u[1] = *(const v4u*)(lb3 + k1 + 16);

        acc0 = __builtin_amdgcn_wmma_f32_16x16x32_bf16(false, fa0.bf, false, b00.bf,
                                                       (short)0, acc0, false, false);
        acc1 = __builtin_amdgcn_wmma_f32_16x16x32_bf16(false, fa0.bf, false, b01.bf,
                                                       (short)0, acc1, false, false);
        acc2 = __builtin_amdgcn_wmma_f32_16x16x32_bf16(false, fa0.bf, false, b02.bf,
                                                       (short)0, acc2, false, false);
        acc3 = __builtin_amdgcn_wmma_f32_16x16x32_bf16(false, fa0.bf, false, b03.bf,
                                                       (short)0, acc3, false, false);

        // Prefetch the next iteration's first A fragment between the two
        // WMMA groups (overlaps the global load with the 2nd group).
        const int kn = k0 + 64;
        if (kn < K) {
            fa0.u[0] = *(const v4u*)(aPtr + kn);
            fa0.u[1] = *(const v4u*)(aPtr + kn + 16);
        }

        acc0 = __builtin_amdgcn_wmma_f32_16x16x32_bf16(false, fa1.bf, false, b10.bf,
                                                       (short)0, acc0, false, false);
        acc1 = __builtin_amdgcn_wmma_f32_16x16x32_bf16(false, fa1.bf, false, b11.bf,
                                                       (short)0, acc1, false, false);
        acc2 = __builtin_amdgcn_wmma_f32_16x16x32_bf16(false, fa1.bf, false, b12.bf,
                                                       (short)0, acc2, false, false);
        acc3 = __builtin_amdgcn_wmma_f32_16x16x32_bf16(false, fa1.bf, false, b13.bf,
                                                       (short)0, acc3, false, false);
    }

    // C/D layout: acc[j] -> row m0 + kgrp*8 + j, col n0 + nt*16 + nh.
    T* cBase = C + (size_t)(m0 + kgrp * 8) * N + n0 + nh;
#pragma unroll
    for (int j = 0; j < 8; ++j) {
        T* rowp = cBase + (size_t)j * N;
        T v0 = (T)fmaxf(acc0[j], 0.0f);
        T v1 = (T)fmaxf(acc1[j], 0.0f);
        T v2 = (T)fmaxf(acc2[j], 0.0f);
        T v3 = (T)fmaxf(acc3[j], 0.0f);
        if constexpr (NT_STORE) {
            __builtin_nontemporal_store(v0, rowp + 0);
            __builtin_nontemporal_store(v1, rowp + 16);
            __builtin_nontemporal_store(v2, rowp + 32);
            __builtin_nontemporal_store(v3, rowp + 48);
        } else {
            rowp[0]  = v0;
            rowp[16] = v1;
            rowp[32] = v2;
            rowp[48] = v3;
        }
    }
}

// ---------------------------------------------------------------------------
// Host-side orchestration.
// ---------------------------------------------------------------------------
extern "C" void kernel_launch(void* const* d_in, const int* in_sizes, int n_in,
                              void* d_out, int out_size, void* d_ws, size_t ws_size,
                              hipStream_t stream) {
    (void)in_sizes; (void)n_in; (void)out_size; (void)ws_size;

    // setup_inputs() dict order (interleaved per hop):
    const float* x_src0 = (const float*)d_in[0];   // [256,   500]
    const float* x_dst0 = (const float*)d_in[1];   // [256,   500]
    const float* x_neg0 = (const float*)d_in[2];   // [1280,  500]
    const float* x_src1 = (const float*)d_in[3];   // [2560,  500]
    const float* x_dst1 = (const float*)d_in[4];   // [2560,  500]
    const float* x_neg1 = (const float*)d_in[5];   // [12800, 500]
    const float* x_src2 = (const float*)d_in[6];   // [25600, 500]
    const float* x_dst2 = (const float*)d_in[7];   // [25600, 500]
    const float* x_neg2 = (const float*)d_in[8];   // [128000,500]
    const float* W0     = (const float*)d_in[9];   // [1000,  256]
    const float* W1     = (const float*)d_in[10];  // [512,   128]

    const int D0 = 500, K0P = 1024, H0 = 256;      // layer-0: K pad 1000->1024
    const int D1 = 256, K1P = 512,  H1 = 128;      // layer-1: K = 512 exactly

    // Scratch allocator (256 B aligned) inside d_ws.
    char* ws = (char*)d_ws;
    size_t off = 0;
    auto balloc = [&](size_t elems) -> __bf16* {
        __bf16* p = (__bf16*)(ws + off);
        off += ((elems * sizeof(__bf16)) + 255) & ~(size_t)255;
        return p;
    };

    __bf16* W0t   = balloc((size_t)H0 * K0P);       // [256, 1024]
    __bf16* W1t   = balloc((size_t)H1 * K1P);       // [128, 512]
    __bf16* A_s0  = balloc((size_t)256   * K0P);    // layer-0 concat inputs
    __bf16* A_d0  = balloc((size_t)256   * K0P);
    __bf16* A_n0  = balloc((size_t)1280  * K0P);
    __bf16* A_s1  = balloc((size_t)2560  * K0P);
    __bf16* A_d1  = balloc((size_t)2560  * K0P);
    __bf16* A_n1  = balloc((size_t)12800 * K0P);
    __bf16* H_s0  = balloc((size_t)256   * H0);     // layer-0 hidden (bf16)
    __bf16* H_d0  = balloc((size_t)256   * H0);
    __bf16* H_n0  = balloc((size_t)1280  * H0);
    __bf16* H_s1  = balloc((size_t)2560  * H0);
    __bf16* H_d1  = balloc((size_t)2560  * H0);
    __bf16* H_n1  = balloc((size_t)12800 * H0);
    __bf16* A2_s  = balloc((size_t)256   * K1P);    // layer-1 concat inputs
    __bf16* A2_d  = balloc((size_t)256   * K1P);
    __bf16* A2_n  = balloc((size_t)1280  * K1P);

    float* out_src = (float*)d_out;                 // [256, 128]
    float* out_dst = out_src + (size_t)256 * H1;    // [256, 128]
    float* out_neg = out_dst + (size_t)256 * H1;    // [1280,128]

    const dim3 blk(256);
    const size_t sh0 = (size_t)64 * (K0P + 8) * sizeof(__bf16);  // ~129 KB
    const size_t sh1 = (size_t)64 * (K1P + 8) * sizeof(__bf16);  // ~65 KB

    // ---- Pack weights to bf16 [N, Kpad] ----
    pack_w_kernel<<<dim3(H0), blk, 0, stream>>>(W0, W0t, 2 * D0, H0, K0P);
    pack_w_kernel<<<dim3(H1), blk, 0, stream>>>(W1, W1t, 2 * D1, H1, K1P);

    // ---- Layer 0: mean-aggregate + concat (f32 -> bf16, NT loads) ----
    agg_concat_f32_kernel<<<dim3(256),   blk, 0, stream>>>(x_src0, x_src1, A_s0, D0, K0P);
    agg_concat_f32_kernel<<<dim3(256),   blk, 0, stream>>>(x_dst0, x_dst1, A_d0, D0, K0P);
    agg_concat_f32_kernel<<<dim3(1280),  blk, 0, stream>>>(x_neg0, x_neg1, A_n0, D0, K0P);
    agg_concat_f32_kernel<<<dim3(2560),  blk, 0, stream>>>(x_src1, x_src2, A_s1, D0, K0P);
    agg_concat_f32_kernel<<<dim3(2560),  blk, 0, stream>>>(x_dst1, x_dst2, A_d1, D0, K0P);
    agg_concat_f32_kernel<<<dim3(12800), blk, 0, stream>>>(x_neg1, x_neg2, A_n1, D0, K0P);

    // ---- Layer 0 GEMMs: [M,1024] @ [1024,256], relu, bf16 out ----
    gemm_bf16_relu_kernel<__bf16, false><<<dim3(256   / 128, H0 / 64), blk, sh0, stream>>>(A_s0, W0t, H_s0, 256,   H0, K0P);
    gemm_bf16_relu_kernel<__bf16, false><<<dim3(256   / 128, H0 / 64), blk, sh0, stream>>>(A_d0, W0t, H_d0, 256,   H0, K0P);
    gemm_bf16_relu_kernel<__bf16, false><<<dim3(1280  / 128, H0 / 64), blk, sh0, stream>>>(A_n0, W0t, H_n0, 1280,  H0, K0P);
    gemm_bf16_relu_kernel<__bf16, false><<<dim3(2560  / 128, H0 / 64), blk, sh0, stream>>>(A_s1, W0t, H_s1, 2560,  H0, K0P);
    gemm_bf16_relu_kernel<__bf16, false><<<dim3(2560  / 128, H0 / 64), blk, sh0, stream>>>(A_d1, W0t, H_d1, 2560,  H0, K0P);
    gemm_bf16_relu_kernel<__bf16, false><<<dim3(12800 / 128, H0 / 64), blk, sh0, stream>>>(A_n1, W0t, H_n1, 12800, H0, K0P);

    // ---- Layer 1: mean-aggregate + concat (bf16 -> bf16) ----
    agg_concat_bf16_kernel<<<dim3(256),  blk, 0, stream>>>(H_s0, H_s1, A2_s, D1, K1P);
    agg_concat_bf16_kernel<<<dim3(256),  blk, 0, stream>>>(H_d0, H_d1, A2_d, D1, K1P);
    agg_concat_bf16_kernel<<<dim3(1280), blk, 0, stream>>>(H_n0, H_n1, A2_n, D1, K1P);

    // ---- Layer 1 GEMMs: [M,512] @ [512,128], relu, f32 NT-stored to d_out --
    gemm_bf16_relu_kernel<float, true><<<dim3(256  / 128, H1 / 64), blk, sh1, stream>>>(A2_s, W1t, out_src, 256,  H1, K1P);
    gemm_bf16_relu_kernel<float, true><<<dim3(256  / 128, H1 / 64), blk, sh1, stream>>>(A2_d, W1t, out_dst, 256,  H1, K1P);
    gemm_bf16_relu_kernel<float, true><<<dim3(1280 / 128, H1 / 64), blk, sh1, stream>>>(A2_n, W1t, out_neg, 1280, H1, K1P);
}
